// MambaSSMBlock_28260884807860
// MI455X (gfx1250) — matrix operand order repaired and checked by reading, not compiled, not run on hardware
//
#include <hip/hip_runtime.h>
#include <hip/hip_bf16.h>

// ---------------------------------------------------------------------------
// Mamba SSM block for gfx1250 (MI455X).
// GEMMs via v_wmma_f32_16x16x32_f16 (f16 in, f32 accumulate), fused epilogues.
// Selective scan: one thread per (batch, channel), h-state in registers,
// B_t/C_t staged in LDS, gating fused.
// ---------------------------------------------------------------------------

#define MB_B       2
#define MB_L       2048
#define MB_DMODEL  1024
#define MB_DINNER  2048
#define MB_DSTATE  16
#define MB_DTRANK  64
#define MB_M       (MB_B * MB_L)          // 4096 rows in every GEMM

typedef __attribute__((ext_vector_type(16))) _Float16 v16h;
typedef __attribute__((ext_vector_type(8)))  _Float16 v8h;
typedef __attribute__((ext_vector_type(8)))  float    v8f;

// ---- WMMA fragment loaders (CDNA5 documented VGPR layouts, wave32) --------

// A fragment 16x32 f16: lane m = lane&15, half h = lane>>4.
// elements 0..7  <- A[m][k0 + 8h .. +7], elements 8..15 <- A[m][k0+16+8h .. +7]
__device__ __forceinline__ v16h load_a_frag(const _Float16* A, int lda,
                                            int m0, int k0, int lane) {
  int m = lane & 15, h = lane >> 4;
  const _Float16* p = A + (size_t)(m0 + m) * lda + k0 + 8 * h;
  v8h lo = *(const v8h*)p;
  v8h hi = *(const v8h*)(p + 16);
  v16h r;
#pragma unroll
  for (int i = 0; i < 8; ++i) { r[i] = lo[i]; r[i + 8] = hi[i]; }
  return r;
}

// B fragment 32x16 f16 from row-major weight W[N][K] (B = W^T):
// lane n = lane&15, half h = lane>>4; elements 0..7 <- W[n][k0+8h .. +7], elements 8..15 <- W[n][k0+16+8h .. +7]
__device__ __forceinline__ v16h load_b_frag(const _Float16* W, int ldw,
                                            int n0, int k0, int lane) {
  int n = lane & 15, h = lane >> 4;
  const _Float16* p = W + (size_t)(n0 + n) * ldw + k0 + 8 * h;
  v8h lo = *(const v8h*)p;
  v8h hi = *(const v8h*)(p + 16);
  v16h r;
#pragma unroll
  for (int i = 0; i < 8; ++i) { r[i] = lo[i]; r[i + 8] = hi[i]; }
  return r;
}

__device__ __forceinline__ v8f wmma_f16(v16h a, v16h b, v8f c) {
  return __builtin_amdgcn_wmma_f32_16x16x32_f16(
      /*neg_a=*/false, a, /*neg_b=*/false, b,
      /*c_mod=*/(short)0, c, /*reuse_a=*/false, /*reuse_b=*/false);
}

// epilogue: 0 = none, 1 = softplus(acc + bias[n]), 2 = silu(acc + bias[n])
__device__ __forceinline__ void store_c_frag(float* C, _Float16* C16,
                                             const float* bias, int ldc,
                                             int m0, int n0, v8f c, int lane,
                                             int epi) {
  int n = lane & 15, h = lane >> 4;
  int col = n0 + n;
  float bv = bias ? bias[col] : 0.f;
#pragma unroll
  for (int r = 0; r < 8; ++r) {
    int row = m0 + r + 8 * h;
    float v = c[r];
    if (epi == 1) { v += bv; v = (v > 20.f) ? v : log1pf(__expf(v)); }
    else if (epi == 2) { v += bv; v = v / (1.f + __expf(-v)); }
    size_t idx = (size_t)row * ldc + col;
    if (C)   C[idx]   = v;
    if (C16) C16[idx] = (_Float16)v;
  }
}

// ---- Generic GEMM: C[M,N] = A[M,K] @ W[N,K]^T, fused epilogue -------------
// Block: 256 threads = 8 waves; wave tile 32x32 (2x2 WMMA); block tile 64x128.
__global__ __launch_bounds__(256) void gemm_f16_wmma_kernel(
    const _Float16* __restrict__ A, const _Float16* __restrict__ W,
    float* __restrict__ C, _Float16* __restrict__ C16,
    const float* __restrict__ bias, int M, int N, int K, int epi) {
  int lane = threadIdx.x & 31;
  int wave = threadIdx.x >> 5;
  int m_base = blockIdx.y * 64 + (wave >> 2) * 32;
  int n_base = blockIdx.x * 128 + (wave & 3) * 32;
  if (m_base >= M) return;
  bool nv0 = n_base < N;          // wave-uniform guards (EXEC all-1s for WMMA)
  bool nv1 = (n_base + 16) < N;
  if (!nv0) return;

  v8f c00 = {}, c01 = {}, c10 = {}, c11 = {};
  for (int k0 = 0; k0 < K; k0 += 32) {
    v16h a0 = load_a_frag(A, K, m_base, k0, lane);
    v16h a1 = load_a_frag(A, K, m_base + 16, k0, lane);
    v16h b0 = load_b_frag(W, K, n_base, k0, lane);
    c00 = wmma_f16(a0, b0, c00);
    c10 = wmma_f16(a1, b0, c10);
    if (nv1) {
      v16h b1 = load_b_frag(W, K, n_base + 16, k0, lane);
      c01 = wmma_f16(a0, b1, c01);
      c11 = wmma_f16(a1, b1, c11);
    }
  }
  store_c_frag(C, C16, bias, N, m_base,      n_base,      c00, lane, epi);
  store_c_frag(C, C16, bias, N, m_base + 16, n_base,      c10, lane, epi);
  if (nv1) {
    store_c_frag(C, C16, bias, N, m_base,      n_base + 16, c01, lane, epi);
    store_c_frag(C, C16, bias, N, m_base + 16, n_base + 16, c11, lane, epi);
  }
}

// ---- f32 -> f16 convert ---------------------------------------------------
__global__ void cvt_f32_f16_kernel(const float* __restrict__ s,
                                   _Float16* __restrict__ d, int n) {
  int i = blockIdx.x * blockDim.x + threadIdx.x;
  if (i < n) d[i] = (_Float16)s[i];
}

// extract dt (first 64 cols of x_dbl[:,96]) as f16 for the dt_proj GEMM
__global__ void extract_dt_kernel(const float* __restrict__ xdbl,
                                  _Float16* __restrict__ dt16, int total) {
  int i = blockIdx.x * blockDim.x + threadIdx.x;
  if (i >= total) return;
  int m = i >> 6, k = i & 63;
  dt16[i] = (_Float16)xdbl[(size_t)m * 96 + k];
}

// ---- causal depthwise conv (width 4) + SiLU; writes f32 and f16 copies ----
__global__ void conv_silu_kernel(const float* __restrict__ xz,
                                 const float* __restrict__ conv_w,
                                 const float* __restrict__ conv_b,
                                 float* __restrict__ xc32,
                                 _Float16* __restrict__ xc16, int total) {
  int idx = blockIdx.x * blockDim.x + threadIdx.x;
  if (idx >= total) return;
  int d = idx % MB_DINNER;
  int t = (idx / MB_DINNER) % MB_L;
  int b = idx / (MB_DINNER * MB_L);
  float acc = conv_b[d];
#pragma unroll
  for (int w = 0; w < 4; ++w) {
    int tt = t - 3 + w;
    if (tt >= 0)
      acc += xz[((size_t)b * MB_L + tt) * (2 * MB_DINNER) + d] * conv_w[d * 4 + w];
  }
  float s = acc / (1.f + __expf(-acc));
  xc32[idx] = s;
  xc16[idx] = (_Float16)s;
}

// ---- selective scan: h_t = exp(dt*A)h + dt*B_t*u ; y = <h,C_t> ------------
// fused: y = (y + D*u) * silu(z), written as f16 for the next GEMM.
__global__ __launch_bounds__(256) void scan_kernel(
    const float* __restrict__ xdbl,  const float* __restrict__ delta,
    const float* __restrict__ xc32,  const float* __restrict__ xz,
    const float* __restrict__ A_log, const float* __restrict__ Dp,
    _Float16* __restrict__ y16) {
  __shared__ float bc[32];                       // B_t[0..15], C_t[16..31]
  const int blocks_per_b = MB_DINNER / 256;      // 8
  int b = blockIdx.x / blocks_per_b;
  int d = (blockIdx.x % blocks_per_b) * 256 + threadIdx.x;

  float Ar[MB_DSTATE];
#pragma unroll
  for (int n = 0; n < MB_DSTATE; ++n)
    Ar[n] = -__expf(A_log[(size_t)d * MB_DSTATE + n]);
  float Dd = Dp[d];
  float h[MB_DSTATE];
#pragma unroll
  for (int n = 0; n < MB_DSTATE; ++n) h[n] = 0.f;

  for (int t = 0; t < MB_L; ++t) {
    size_t row = (size_t)b * MB_L + t;
    __syncthreads();
    if (threadIdx.x < 32) bc[threadIdx.x] = xdbl[row * 96 + 64 + threadIdx.x];
    __syncthreads();
    float dt = delta[row * MB_DINNER + d];
    float u  = xc32[row * MB_DINNER + d];
    float zv = xz[row * (2 * MB_DINNER) + MB_DINNER + d];
    float du = dt * u;
    float y = 0.f;
#pragma unroll
    for (int n = 0; n < MB_DSTATE; ++n) {
      h[n] = __expf(dt * Ar[n]) * h[n] + du * bc[n];
      y += h[n] * bc[16 + n];
    }
    y = (y + Dd * u) * (zv / (1.f + __expf(-zv)));
    y16[row * MB_DINNER + d] = (_Float16)y;
  }
}

// ---------------------------------------------------------------------------
static inline char* wsalloc(char*& p, size_t bytes) {
  char* r = p;
  p += (bytes + 255) & ~(size_t)255;
  return r;
}

extern "C" void kernel_launch(void* const* d_in, const int* in_sizes, int n_in,
                              void* d_out, int out_size, void* d_ws, size_t ws_size,
                              hipStream_t stream) {
  (void)in_sizes; (void)n_in; (void)out_size; (void)ws_size;
  const float* x         = (const float*)d_in[0];   // (B,L,1024)
  const float* in_proj_w = (const float*)d_in[1];   // (4096,1024)
  const float* conv_w    = (const float*)d_in[2];   // (2048,4)
  const float* conv_b    = (const float*)d_in[3];   // (2048)
  const float* x_proj_w  = (const float*)d_in[4];   // (96,2048)
  const float* dt_proj_w = (const float*)d_in[5];   // (2048,64)
  const float* dt_proj_b = (const float*)d_in[6];   // (2048)
  const float* A_log     = (const float*)d_in[7];   // (2048,16)
  const float* Dp        = (const float*)d_in[8];   // (2048)
  const float* ssm_out_w = (const float*)d_in[9];   // (1024,2048)
  const float* final_w   = (const float*)d_in[10];  // (1024,1024)
  const float* final_b   = (const float*)d_in[11];  // (1024)
  float* out_f = (float*)d_out;                     // (B,L,1024) f32

  char* p = (char*)d_ws;
  _Float16* w_in16  = (_Float16*)wsalloc(p, (size_t)4096 * 1024 * 2);
  _Float16* w_xp16  = (_Float16*)wsalloc(p, (size_t)96   * 2048 * 2);
  _Float16* w_dt16  = (_Float16*)wsalloc(p, (size_t)2048 * 64   * 2);
  _Float16* w_out16 = (_Float16*)wsalloc(p, (size_t)1024 * 2048 * 2);
  _Float16* w_fin16 = (_Float16*)wsalloc(p, (size_t)1024 * 1024 * 2);
  _Float16* x16     = (_Float16*)wsalloc(p, (size_t)MB_M * 1024 * 2);
  float*    xz      = (float*)   wsalloc(p, (size_t)MB_M * 4096 * 4);
  float*    xc32    = (float*)   wsalloc(p, (size_t)MB_M * 2048 * 4);
  _Float16* xc16    = (_Float16*)wsalloc(p, (size_t)MB_M * 2048 * 2);
  float*    xdbl    = (float*)   wsalloc(p, (size_t)MB_M * 96   * 4);
  _Float16* dt16    = (_Float16*)wsalloc(p, (size_t)MB_M * 64   * 2);
  float*    delta   = (float*)   wsalloc(p, (size_t)MB_M * 2048 * 4);
  _Float16* y16     = (_Float16*)wsalloc(p, (size_t)MB_M * 2048 * 2);
  _Float16* out16   = (_Float16*)wsalloc(p, (size_t)MB_M * 1024 * 2);

  auto cdiv = [](int a, int b) { return (a + b - 1) / b; };

  // weight + activation converts to f16
  cvt_f32_f16_kernel<<<cdiv(4096 * 1024, 256), 256, 0, stream>>>(in_proj_w, w_in16, 4096 * 1024);
  cvt_f32_f16_kernel<<<cdiv(96 * 2048, 256),   256, 0, stream>>>(x_proj_w,  w_xp16, 96 * 2048);
  cvt_f32_f16_kernel<<<cdiv(2048 * 64, 256),   256, 0, stream>>>(dt_proj_w, w_dt16, 2048 * 64);
  cvt_f32_f16_kernel<<<cdiv(1024 * 2048, 256), 256, 0, stream>>>(ssm_out_w, w_out16, 1024 * 2048);
  cvt_f32_f16_kernel<<<cdiv(1024 * 1024, 256), 256, 0, stream>>>(final_w,   w_fin16, 1024 * 1024);
  cvt_f32_f16_kernel<<<cdiv(MB_M * 1024, 256), 256, 0, stream>>>(x,         x16,    MB_M * 1024);

  // 1) xz = x @ in_proj_w^T            (4096 x 4096 x 1024)
  gemm_f16_wmma_kernel<<<dim3(cdiv(4096, 128), MB_M / 64), 256, 0, stream>>>(
      x16, w_in16, xz, nullptr, nullptr, MB_M, 4096, 1024, 0);

  // 2) causal depthwise conv + SiLU -> xc (f32 + f16)
  conv_silu_kernel<<<cdiv(MB_M * 2048, 256), 256, 0, stream>>>(
      xz, conv_w, conv_b, xc32, xc16, MB_M * 2048);

  // 3) x_dbl = xc @ x_proj_w^T         (4096 x 96 x 2048)
  gemm_f16_wmma_kernel<<<dim3(cdiv(96, 128), MB_M / 64), 256, 0, stream>>>(
      xc16, w_xp16, xdbl, nullptr, nullptr, MB_M, 96, 2048, 0);

  // 4) delta = softplus(dt @ dt_proj_w^T + dt_proj_b)   (4096 x 2048 x 64)
  extract_dt_kernel<<<cdiv(MB_M * 64, 256), 256, 0, stream>>>(xdbl, dt16, MB_M * 64);
  gemm_f16_wmma_kernel<<<dim3(cdiv(2048, 128), MB_M / 64), 256, 0, stream>>>(
      dt16, w_dt16, delta, nullptr, dt_proj_b, MB_M, 2048, 64, 1);

  // 5) selective scan + D-skip + SiLU(z) gating -> y (f16)
  scan_kernel<<<MB_B * (MB_DINNER / 256), 256, 0, stream>>>(
      xdbl, delta, xc32, xz, A_log, Dp, y16);

  // 6) out = y @ ssm_out_w^T           (4096 x 1024 x 2048), f16 shadow only
  gemm_f16_wmma_kernel<<<dim3(cdiv(1024, 128), MB_M / 64), 256, 0, stream>>>(
      y16, w_out16, nullptr, out16, nullptr, MB_M, 1024, 2048, 0);

  // 7) final = silu(out @ final_w^T + final_b) -> d_out (f32)
  gemm_f16_wmma_kernel<<<dim3(cdiv(1024, 128), MB_M / 64), 256, 0, stream>>>(
      out16, w_fin16, out_f, nullptr, final_b, MB_M, 1024, 1024, 2);
}
